// FlashGPTJAttention_51470888075971
// MI455X (gfx1250) — matrix operand order repaired and checked by reading, not compiled, not run on hardware
//
#include <hip/hip_runtime.h>
#include <hip/hip_bf16.h>
#include <cstdint>
#include <cstddef>

// ---------------- problem constants (from reference) ----------------
#define NH        16
#define HD        256
#define HID       4096
#define QKV_LD    12288      // 3*HID
#define TOKENS    8192
#define BATCH     8
#define SEQ       1024       // TOKENS/BATCH
#define ROT_HALF  32         // ROTARY_DIM/2

typedef __bf16 bf16_t;
typedef __attribute__((ext_vector_type(16))) __bf16 v16bf;
typedef __attribute__((ext_vector_type(2)))  __bf16 bf16x2;
typedef __attribute__((ext_vector_type(8)))  float  v8f;
typedef __attribute__((ext_vector_type(4)))  unsigned int v4u;
typedef __attribute__((ext_vector_type(8)))  int v8i;
typedef __attribute__((ext_vector_type(4)))  int v4i;

static __device__ inline v8f zero8() {
  v8f z;
#pragma unroll
  for (int i = 0; i < 8; ++i) z[i] = 0.0f;
  return z;
}

static __device__ inline v8f wmma_bf16(v16bf a, v16bf b, v8f c) {
  return __builtin_amdgcn_wmma_f32_16x16x32_bf16(
      /*neg_a=*/false, a, /*neg_b=*/false, b,
      /*c_mod=*/(short)0, c, /*reuse_a=*/false, /*reuse_b=*/false);
}

// A fragment, 16x32 bf16 (M x K). `A` points at row0 of the tile.
// Lane L: m=L&15, hg=L>>4. VGPR r<4: K = 8*hg + 2r,+1 ; r>=4: K = 16 + 8*hg + 2(r-4),+1.
static __device__ inline v16bf load_frag_a(const bf16_t* __restrict__ A, int lda, int kk) {
  const int lane = threadIdx.x & 31;
  const int m  = lane & 15;
  const int hg = lane >> 4;
  const bf16_t* base = A + (size_t)m * (size_t)lda + (size_t)(kk + 8 * hg);
  v16bf a;
#pragma unroll
  for (int r = 0; r < 8; ++r) {
    const int off = (r < 4) ? (2 * r) : (8 + 2 * r);
    bf16x2 p = *reinterpret_cast<const bf16x2*>(base + off);
    a[2 * r]     = p[0];
    a[2 * r + 1] = p[1];
  }
  return a;
}

// B fragment, 32x16 bf16 (K x N), where B[k][n] = S[n][k] (S row-major, ld).
// `S` points at row col0 of S. Lane L: n=L&15, hg=L>>4. VGPR r<4: K = 8*hg + 2r,+1 ; r>=4: K = 16 + 8*hg + 2(r-4),+1.
static __device__ inline v16bf load_frag_b(const bf16_t* __restrict__ S, int ld, int kk) {
  const int lane = threadIdx.x & 31;
  const int n  = lane & 15;
  const int hg = lane >> 4;
  const bf16_t* base = S + (size_t)n * (size_t)ld + (size_t)(kk + 8 * hg);
  v16bf b;
#pragma unroll
  for (int r = 0; r < 8; ++r) {
    const int off = (r < 4) ? (2 * r) : (8 + 2 * r);
    bf16x2 p = *reinterpret_cast<const bf16x2*>(base + off);
    b[2 * r]     = p[0];
    b[2 * r + 1] = p[1];
  }
  return b;
}

// ---------------- Tensor Data Mover (TDM) 2D tile load ----------------
#if __has_builtin(__builtin_amdgcn_tensor_load_to_lds)
#define USE_TDM 1
#else
#define USE_TDM 0
#endif

#if USE_TDM
// LDS byte offset of a __shared__ object: generic LDS addresses are
// {aperture_hi, 32-bit workgroup-relative offset} -> low 32 bits.
static __device__ inline unsigned int lds_off(const void* p) {
  return (unsigned int)(uintptr_t)p;
}

// 2D tile DMA: tile_d0 elements (bf16) per row, tile_d1 rows, global row pitch
// `stride0` elements. Tensor dims == tile dims (tilings divide evenly -> no OOB).
// TDM LDS padding: after every 16 DWORDs (64B = one 32-elem row) insert 4 DWORDs
// (16B) -> LDS row pitch 40 elements, which de-conflicts the 16 row-lanes of a
// WMMA fragment across the 64 LDS banks.
static __device__ inline void tdm_load_2d_pad(unsigned int lds_addr, const bf16_t* gptr,
                                              unsigned int tile_d0, unsigned int tile_d1,
                                              unsigned int stride0) {
  const unsigned long long ga = (unsigned long long)(uintptr_t)gptr;
  v4u g0;
  g0[0] = 1u;                                   // count=1, user mode, no gather
  g0[1] = lds_addr;                             // LDS byte address
  g0[2] = (unsigned int)ga;                     // global_addr[31:0]
  g0[3] = (unsigned int)(ga >> 32) | (2u << 30);// global_addr[56:32] | type=2
  v8i g1;
  g1[0] = (int)((1u << 16)                      // data_size = 2 bytes
              | (1u << 20)                      // pad_enable
              | (3u << 22)                      // pad_interval: 16 DWORDs (64B)
              | (3u << 25));                    // pad_amount:   4 DWORDs (16B)
  g1[1] = (int)((tile_d0 & 0xFFFFu) << 16);                         // tensor_dim0 lo
  g1[2] = (int)((tile_d0 >> 16) | ((tile_d1 & 0xFFFFu) << 16));     // td0 hi | tensor_dim1 lo
  g1[3] = (int)((tile_d1 >> 16) | (tile_d0 << 16));                 // td1 hi | tile_dim0
  g1[4] = (int)(tile_d1 & 0xFFFFu);                                 // tile_dim1 (tile_dim2=0)
  g1[5] = (int)stride0;                                             // tensor_dim0_stride lo
  g1[6] = 0;
  g1[7] = 0;
  v4i z4 = {0, 0, 0, 0};
#if __clang_major__ >= 23
  v8i z8 = {0, 0, 0, 0, 0, 0, 0, 0};
  __builtin_amdgcn_tensor_load_to_lds(g0, g1, z4, z4, z8, 0);
#else
  __builtin_amdgcn_tensor_load_to_lds(g0, g1, z4, z4, 0);
#endif
}
#endif  // USE_TDM

// ---------------- fp32 -> bf16 convert ----------------
__global__ void cvt_f32_bf16(const float* __restrict__ in, bf16_t* __restrict__ out, long n) {
  long i = (long)blockIdx.x * blockDim.x + threadIdx.x;
  if (i < n) out[i] = (bf16_t)in[i];
}

// ---------------- GEMM: C[M,N] = A[M,K] * W[N,K]^T (bf16 in, f32 accum) ----------------
// block: 256 threads = 8 waves in 2(M) x 4(N); block tile 64x256; wave tile 32x64.
// TDM path: K-step 32 tiles double-buffered in LDS (row pitch 40 elems via TDM pad),
// each wave DMAs an 8-row A slice + 32-row B slice, s_wait_tensorcnt + barrier per step.
#define LDS_PITCH 40

template <bool OUT_BF16>
__global__ __launch_bounds__(256)
void gemm_bf16_nt(const bf16_t* __restrict__ A, const bf16_t* __restrict__ W,
                  void* __restrict__ Cout, int K, int ldc) {
  const int wave = threadIdx.x >> 5;
  const int lane = threadIdx.x & 31;
  const int wm = wave >> 2;          // 0..1
  const int wn = wave & 3;           // 0..3
  const int row0 = blockIdx.y * 64 + wm * 32;
  const int col0 = blockIdx.x * 256 + wn * 64;

  v8f acc[2][4];
#pragma unroll
  for (int i = 0; i < 2; ++i)
#pragma unroll
    for (int j = 0; j < 4; ++j) acc[i][j] = zero8();

#if USE_TDM
  __shared__ __align__(16) bf16_t sA[2][64 * LDS_PITCH];    // 10 KiB
  __shared__ __align__(16) bf16_t sB[2][256 * LDS_PITCH];   // 40 KiB
  const int row0g = blockIdx.y * 64;
  const int col0g = blockIdx.x * 256;
  const int wave_u = __builtin_amdgcn_readfirstlane(threadIdx.x >> 5);
  const int nsteps = K >> 5;

  // prologue: fill buffer 0
  tdm_load_2d_pad(lds_off(&sA[0][0]) + (unsigned)(wave_u * 8) * LDS_PITCH * 2,
                  A + (size_t)(row0g + wave_u * 8) * (size_t)K, 32, 8, (unsigned)K);
  tdm_load_2d_pad(lds_off(&sB[0][0]) + (unsigned)(wave_u * 32) * LDS_PITCH * 2,
                  W + (size_t)(col0g + wave_u * 32) * (size_t)K, 32, 32, (unsigned)K);

  for (int s = 0; s < nsteps; ++s) {
    const int cur = s & 1;
    __builtin_amdgcn_s_wait_tensorcnt(0);
    __syncthreads();
    if (s + 1 < nsteps) {
      const int nxt = cur ^ 1;
      const int kk = (s + 1) << 5;
      tdm_load_2d_pad(lds_off(&sA[nxt][0]) + (unsigned)(wave_u * 8) * LDS_PITCH * 2,
                      A + (size_t)(row0g + wave_u * 8) * (size_t)K + kk, 32, 8, (unsigned)K);
      tdm_load_2d_pad(lds_off(&sB[nxt][0]) + (unsigned)(wave_u * 32) * LDS_PITCH * 2,
                      W + (size_t)(col0g + wave_u * 32) * (size_t)K + kk, 32, 32, (unsigned)K);
    }
    v16bf af[2];
#pragma unroll
    for (int i = 0; i < 2; ++i)
      af[i] = load_frag_a(&sA[cur][(wm * 32 + 16 * i) * LDS_PITCH], LDS_PITCH, 0);
    v16bf bfr[4];
#pragma unroll
    for (int j = 0; j < 4; ++j)
      bfr[j] = load_frag_b(&sB[cur][(wn * 64 + 16 * j) * LDS_PITCH], LDS_PITCH, 0);
#pragma unroll
    for (int i = 0; i < 2; ++i)
#pragma unroll
      for (int j = 0; j < 4; ++j)
        acc[i][j] = wmma_bf16(af[i], bfr[j], acc[i][j]);
  }
#else
  for (int kk = 0; kk < K; kk += 32) {
    v16bf af[2];
#pragma unroll
    for (int i = 0; i < 2; ++i)
      af[i] = load_frag_a(A + (size_t)(row0 + 16 * i) * (size_t)K, K, kk);
    v16bf bfr[4];
#pragma unroll
    for (int j = 0; j < 4; ++j)
      bfr[j] = load_frag_b(W + (size_t)(col0 + 16 * j) * (size_t)K, K, kk);
#pragma unroll
    for (int i = 0; i < 2; ++i)
#pragma unroll
      for (int j = 0; j < 4; ++j)
        acc[i][j] = wmma_bf16(af[i], bfr[j], acc[i][j]);
  }
#endif

  const int n  = lane & 15;
  const int hg = lane >> 4;
#pragma unroll
  for (int i = 0; i < 2; ++i)
#pragma unroll
    for (int j = 0; j < 4; ++j)
#pragma unroll
      for (int r = 0; r < 8; ++r) {
        const int row = row0 + 16 * i + r + 8 * hg;
        const int col = col0 + 16 * j + n;
        const size_t idx = (size_t)row * (size_t)ldc + col;
        if (OUT_BF16) ((bf16_t*)Cout)[idx] = (bf16_t)acc[i][j][r];
        else          ((float*)Cout)[idx]  = acc[i][j][r];
      }
}

// ---------------- RoPE in-place on q,k inside qkv (bf16) ----------------
__global__ void rope_kernel(bf16_t* __restrict__ qkv,
                            const float* __restrict__ cosp,
                            const float* __restrict__ sinp) {
  const long i = (long)blockIdx.x * blockDim.x + threadIdx.x;  // TOKENS*NH*32
  const int d = (int)(i & 31);
  const int h = (int)((i >> 5) & (NH - 1));
  const long t = i >> 9;
  if (t >= TOKENS) return;
  const float c = cosp[t * ROT_HALF + d];
  const float s = sinp[t * ROT_HALF + d];
  {
    const size_t base = (size_t)t * QKV_LD + (size_t)h * HD;
    const float x1 = (float)qkv[base + d];
    const float x2 = (float)qkv[base + ROT_HALF + d];
    qkv[base + d]            = (bf16_t)(x1 * c - x2 * s);
    qkv[base + ROT_HALF + d] = (bf16_t)(x2 * c + x1 * s);
  }
  {
    const size_t base = (size_t)t * QKV_LD + HID + (size_t)h * HD;
    const float x1 = (float)qkv[base + d];
    const float x2 = (float)qkv[base + ROT_HALF + d];
    qkv[base + d]            = (bf16_t)(x1 * c - x2 * s);
    qkv[base + ROT_HALF + d] = (bf16_t)(x2 * c + x1 * s);
  }
}

// ---------------- V transpose: vt[(b,h)][dim][key] = V[key][dim] ----------------
__global__ void build_vt(const bf16_t* __restrict__ qkv, bf16_t* __restrict__ vt) {
  const long i = (long)blockIdx.x * blockDim.x + threadIdx.x;  // BATCH*NH*HD*SEQ
  const int srow = (int)(i & (SEQ - 1));
  const int dim  = (int)((i >> 10) & (HD - 1));
  const int bh   = (int)(i >> 18);
  const int b = bh >> 4, h = bh & (NH - 1);
  const long t = (long)b * SEQ + srow;
  vt[i] = qkv[(size_t)t * QKV_LD + 2 * HID + (size_t)h * HD + dim];
}

// ---------------- Flash attention (causal, online softmax) ----------------
// grid: (SEQ/64, BATCH*NH); block: 128 threads = 4 waves; wave handles 16 query rows.
__global__ __launch_bounds__(128)
void flash_attn(const bf16_t* __restrict__ qkv, const bf16_t* __restrict__ vt,
                bf16_t* __restrict__ aout) {
  __shared__ __align__(16) bf16_t pstage[4][16 * 32];

  const int wave = threadIdx.x >> 5;
  const int lane = threadIdx.x & 31;
  const int n  = lane & 15;
  const int hg = lane >> 4;
  const int bh = blockIdx.y;
  const int b = bh >> 4, h = bh & (NH - 1);
  const int q0 = blockIdx.x * 64 + wave * 16;

  const bf16_t* Q  = qkv + (size_t)((size_t)b * SEQ + q0) * QKV_LD + (size_t)h * HD;
  const bf16_t* Kp = qkv + (size_t)((size_t)b * SEQ) * QKV_LD + HID + (size_t)h * HD;
  const bf16_t* Vt = vt + (size_t)bh * HD * SEQ;

  v16bf qf[8];
#pragma unroll
  for (int c = 0; c < 8; ++c) qf[c] = load_frag_a(Q, QKV_LD, c * 32);

  v8f acc[16];
#pragma unroll
  for (int f = 0; f < 16; ++f) acc[f] = zero8();
  float rm[8], rl[8];
#pragma unroll
  for (int r = 0; r < 8; ++r) { rm[r] = -3.0e38f; rl[r] = 0.0f; }

  const float scale = 0.0625f;  // 1/sqrt(256)

  for (int k0 = 0; k0 < q0 + 16; k0 += 32) {
    v8f s0 = zero8(), s1 = zero8();
#pragma unroll
    for (int c = 0; c < 8; ++c) {
      v16bf kb0 = load_frag_b(Kp + (size_t)k0 * QKV_LD,        QKV_LD, c * 32);
      v16bf kb1 = load_frag_b(Kp + (size_t)(k0 + 16) * QKV_LD, QKV_LD, c * 32);
      s0 = wmma_bf16(qf[c], kb0, s0);
      s1 = wmma_bf16(qf[c], kb1, s1);
    }

#pragma unroll
    for (int r = 0; r < 8; ++r) {
      const int row = q0 + r + 8 * hg;
      float a0 = s0[r] * scale;
      float a1 = s1[r] * scale;
      if (k0 + n > row)      a0 = -3.0e38f;
      if (k0 + 16 + n > row) a1 = -3.0e38f;

      float mx = fmaxf(a0, a1);
#pragma unroll
      for (int msk = 8; msk >= 1; msk >>= 1) mx = fmaxf(mx, __shfl_xor(mx, msk, 32));
      const float mnew  = fmaxf(rm[r], mx);
      const float alpha = __expf(rm[r] - mnew);
      const float p0 = __expf(a0 - mnew);
      const float p1 = __expf(a1 - mnew);
      float rs = p0 + p1;
#pragma unroll
      for (int msk = 8; msk >= 1; msk >>= 1) rs += __shfl_xor(rs, msk, 32);
      rl[r] = rl[r] * alpha + rs;
      rm[r] = mnew;
#pragma unroll
      for (int f = 0; f < 16; ++f) acc[f][r] *= alpha;

      const int mrow = r + 8 * hg;
      pstage[wave][mrow * 32 + n]      = (bf16_t)p0;
      pstage[wave][mrow * 32 + 16 + n] = (bf16_t)p1;
    }

    v16bf pf = load_frag_a(&pstage[wave][0], 32, 0);
#pragma unroll
    for (int f = 0; f < 16; ++f) {
      v16bf vb = load_frag_b(Vt + (size_t)(f * 16) * SEQ, SEQ, k0);
      acc[f] = wmma_bf16(pf, vb, acc[f]);
    }
  }

#pragma unroll
  for (int f = 0; f < 16; ++f)
#pragma unroll
    for (int r = 0; r < 8; ++r) {
      const int row = q0 + r + 8 * hg;
      const int col = f * 16 + n;
      aout[(size_t)((size_t)b * SEQ + row) * HID + (size_t)h * HD + col] =
          (bf16_t)(acc[f][r] / rl[r]);
    }
}

// ---------------- host side ----------------
extern "C" void kernel_launch(void* const* d_in, const int* in_sizes, int n_in,
                              void* d_out, int out_size, void* d_ws, size_t ws_size,
                              hipStream_t stream) {
  (void)in_sizes; (void)n_in; (void)out_size; (void)ws_size;
  const float* hidden = (const float*)d_in[0];
  const float* w_qkv  = (const float*)d_in[1];
  const float* w_o    = (const float*)d_in[2];
  const float* cosp   = (const float*)d_in[3];
  const float* sinp   = (const float*)d_in[4];
  // d_in[5..8]: key_cache / value_cache / slots / batch_size — dead w.r.t. returned output
  float* out = (float*)d_out;

  // workspace layout (512 MiB total)
  char* ws = (char*)d_ws;
  bf16_t* hb   = (bf16_t*)ws;  ws += (size_t)TOKENS * HID * 2;          // 64 MiB
  bf16_t* wqb  = (bf16_t*)ws;  ws += (size_t)3 * HID * HID * 2;        // 96 MiB
  bf16_t* wob  = (bf16_t*)ws;  ws += (size_t)HID * HID * 2;            // 32 MiB
  bf16_t* qkvb = (bf16_t*)ws;  ws += (size_t)TOKENS * QKV_LD * 2;      // 192 MiB
  bf16_t* vtb  = (bf16_t*)ws;  ws += (size_t)BATCH * NH * HD * SEQ * 2;// 64 MiB
  bf16_t* aob  = (bf16_t*)ws;                                          // 64 MiB

  {
    long n = (long)TOKENS * HID;
    cvt_f32_bf16<<<(unsigned)((n + 255) / 256), 256, 0, stream>>>(hidden, hb, n);
    n = (long)3 * HID * HID;
    cvt_f32_bf16<<<(unsigned)((n + 255) / 256), 256, 0, stream>>>(w_qkv, wqb, n);
    n = (long)HID * HID;
    cvt_f32_bf16<<<(unsigned)((n + 255) / 256), 256, 0, stream>>>(w_o, wob, n);
  }

  {
    dim3 grid(QKV_LD / 256, TOKENS / 64);
    gemm_bf16_nt<true><<<grid, 256, 0, stream>>>(hb, wqb, (void*)qkvb, HID, QKV_LD);
  }

  {
    long n = (long)TOKENS * NH * ROT_HALF;
    rope_kernel<<<(unsigned)((n + 255) / 256), 256, 0, stream>>>(qkvb, cosp, sinp);
  }

  {
    long n = (long)BATCH * NH * HD * SEQ;
    build_vt<<<(unsigned)((n + 255) / 256), 256, 0, stream>>>(qkvb, vtb);
  }

  {
    dim3 grid(SEQ / 64, BATCH * NH);
    flash_attn<<<grid, 128, 0, stream>>>(qkvb, vtb, aob);
  }

  {
    dim3 grid(HID / 256, TOKENS / 64);
    gemm_bf16_nt<false><<<grid, 256, 0, stream>>>(aob, wob, (void*)out, HID, HID);
  }
}
